// BayesianDenseLayer_45114336477600
// MI455X (gfx1250) — matrix operand order repaired and checked by reading, not compiled, not run on hardware
//
#include <hip/hip_runtime.h>

// Problem sizes (fixed by the reference).
#define D_IN  2048   // K
#define D_OUT 2048   // N
#define BATCH 4096   // M

typedef _Float16 v4h  __attribute__((ext_vector_type(4)));
typedef _Float16 v8h  __attribute__((ext_vector_type(8)));
typedef _Float16 v16h __attribute__((ext_vector_type(16)));
typedef float    v4f  __attribute__((ext_vector_type(4)));
typedef float    v8f  __attribute__((ext_vector_type(8)));
typedef int      v4i  __attribute__((ext_vector_type(4)));

__device__ __forceinline__ float softplus_f(float v) {
    // numerically-stable softplus
    return (v > 20.0f) ? v : log1pf(__expf(v));
}

// ---------------------------------------------------------------------------
// Prep 1: convert x -> f16 and (x * s) -> f16, row-major [BATCH, D_IN].
// Vectorized x4: float4/int4 in, 64-bit f16 stores out (fully coalesced).
// ---------------------------------------------------------------------------
__global__ void flipout_prep_x(const float* __restrict__ x,
                               const int*   __restrict__ s,
                               _Float16* __restrict__ xh,
                               _Float16* __restrict__ xsh) {
    size_t i = ((size_t)blockIdx.x * blockDim.x + threadIdx.x) * 4;
    v4f xv = *(const v4f*)(x + i);
    v4i sv = *(const v4i*)(s + i);
    v4h a, b;
    #pragma unroll
    for (int j = 0; j < 4; ++j) {
        a[j] = (_Float16)xv[j];
        b[j] = (_Float16)(xv[j] * (float)sv[j]);
    }
    *(v4h*)(xh + i)  = a;
    *(v4h*)(xsh + i) = b;
}

// ---------------------------------------------------------------------------
// Prep 2: build f16 *transposed* (N-major) copies of w_loc and
// w_samples = softplus(w_std) * eps_w via LDS tile transpose.
// 32x32 tile per block: reads coalesced along N, writes coalesced along K.
// Padded LDS rows (33) avoid bank conflicts on the transposed read.
// Naive strided f16 stores would amplify 16 MB of writes into ~0.5-1 GB of
// HBM traffic (one cacheline per 2-byte store); tiling removes that.
// ---------------------------------------------------------------------------
__global__ __launch_bounds__(256)
void flipout_prep_w(const float* __restrict__ w_loc,
                    const float* __restrict__ w_std,
                    const float* __restrict__ eps_w,
                    _Float16* __restrict__ wlT,
                    _Float16* __restrict__ wsT) {
    __shared__ _Float16 tl[32][33];
    __shared__ _Float16 ts[32][33];

    const int tx = threadIdx.x & 31;     // fast dim
    const int ty = threadIdx.x >> 5;     // 0..7
    const int kTile = blockIdx.y * 32;
    const int nTile = blockIdx.x * 32;

    #pragma unroll
    for (int r = 0; r < 4; ++r) {
        const int kr = ty + r * 8;               // 0..31
        const size_t i = (size_t)(kTile + kr) * D_OUT + (nTile + tx);
        const float wl = w_loc[i];
        const float wsamp = softplus_f(w_std[i]) * eps_w[i];
        tl[kr][tx] = (_Float16)wl;
        ts[kr][tx] = (_Float16)wsamp;
    }
    __syncthreads();
    #pragma unroll
    for (int r = 0; r < 4; ++r) {
        const int nr = ty + r * 8;               // 0..31
        const size_t t = (size_t)(nTile + nr) * D_IN + (kTile + tx);
        wlT[t] = tl[tx][nr];
        wsT[t] = ts[tx][nr];
    }
}

// ---------------------------------------------------------------------------
// Prep 3: bsamp[n] = softplus(b_std[n]) * eps_b[n]
// ---------------------------------------------------------------------------
__global__ void flipout_prep_b(const float* __restrict__ b_std,
                               const float* __restrict__ eps_b,
                               float* __restrict__ bsamp) {
    int n = blockIdx.x * blockDim.x + threadIdx.x;
    if (n < D_OUT) bsamp[n] = softplus_f(b_std[n]) * eps_b[n];
}

// ---------------------------------------------------------------------------
// Fused dual-GEMM + flipout epilogue.
//   accL = x @ w_loc ; accP = (x*s) @ w_samples
//   out  = accL + r1*accP + b_loc + r2*bsamp
// Block = 256 threads = 8 waves; block tile = 128(M) x 64(N);
// wave tile = 32x32 = 2x2 WMMA tiles, two accumulation streams.
// Working set (48 MB f16) fits the 192 MB L2, so hot-loop loads are L2-fed.
// ---------------------------------------------------------------------------
__global__ __launch_bounds__(256)
void flipout_gemm_wmma(const _Float16* __restrict__ xh,
                       const _Float16* __restrict__ xsh,
                       const _Float16* __restrict__ wlT,
                       const _Float16* __restrict__ wsT,
                       const int*   __restrict__ r1,
                       const int*   __restrict__ r2,
                       const float* __restrict__ b_loc,
                       const float* __restrict__ bsamp,
                       float* __restrict__ out) {
    const int lane  = threadIdx.x & 31;
    const int wave  = threadIdx.x >> 5;   // 0..7
    const int wm    = wave & 3;           // M sub-tile index (0..3)
    const int wn    = wave >> 2;          // N sub-tile index (0..1)

    const int mBase = blockIdx.y * 128 + wm * 32;
    const int nBase = blockIdx.x * 64  + wn * 32;

    const int mrow  = lane & 15;          // row (A) / column (B) within a tile
    const int khalf = lane >> 4;          // 0 or 1: which K-half this lane holds

    // A-fragment base pointers (two K-chunks per lane: +khalf*16 and +khalf*16+8)
    const _Float16* aL0 = xh  + (size_t)(mBase + mrow)      * D_IN + khalf * 16;
    const _Float16* aL1 = xh  + (size_t)(mBase + 16 + mrow) * D_IN + khalf * 16;
    const _Float16* aP0 = xsh + (size_t)(mBase + mrow)      * D_IN + khalf * 16;
    const _Float16* aP1 = xsh + (size_t)(mBase + 16 + mrow) * D_IN + khalf * 16;

    // B-fragment base pointers (transposed weights: column mrow, contiguous K)
    const _Float16* bL0 = wlT + (size_t)(nBase + mrow)      * D_IN + khalf * 16;
    const _Float16* bL1 = wlT + (size_t)(nBase + 16 + mrow) * D_IN + khalf * 16;
    const _Float16* bS0 = wsT + (size_t)(nBase + mrow)      * D_IN + khalf * 16;
    const _Float16* bS1 = wsT + (size_t)(nBase + 16 + mrow) * D_IN + khalf * 16;

    v8f accL[2][2];
    v8f accP[2][2];
    #pragma unroll
    for (int i = 0; i < 2; ++i)
        #pragma unroll
        for (int j = 0; j < 2; ++j) { accL[i][j] = (v8f){}; accP[i][j] = (v8f){}; }

    #pragma unroll 2
    for (int kb = 0; kb < D_IN; kb += 32) {
        // Prefetch the streamed operands a bit ahead (global_prefetch_b8).
        __builtin_prefetch(bL0 + kb + 128, 0, 1);
        __builtin_prefetch(bS0 + kb + 128, 0, 1);

        // ---- A fragments: two b128 loads each, merged into v16h ----
        v8h lo, hi;
        lo = *(const v8h*)(aL0 + kb); hi = *(const v8h*)(aL0 + kb + 8);
        v16h a0 = __builtin_shufflevector(lo, hi, 0,1,2,3,4,5,6,7,8,9,10,11,12,13,14,15);
        lo = *(const v8h*)(aL1 + kb); hi = *(const v8h*)(aL1 + kb + 8);
        v16h a1 = __builtin_shufflevector(lo, hi, 0,1,2,3,4,5,6,7,8,9,10,11,12,13,14,15);
        lo = *(const v8h*)(aP0 + kb); hi = *(const v8h*)(aP0 + kb + 8);
        v16h p0 = __builtin_shufflevector(lo, hi, 0,1,2,3,4,5,6,7,8,9,10,11,12,13,14,15);
        lo = *(const v8h*)(aP1 + kb); hi = *(const v8h*)(aP1 + kb + 8);
        v16h p1 = __builtin_shufflevector(lo, hi, 0,1,2,3,4,5,6,7,8,9,10,11,12,13,14,15);

        // ---- B fragments: 16 contiguous K values per lane ----
        v16h b0 = *(const v16h*)(bL0 + kb);
        v16h b1 = *(const v16h*)(bL1 + kb);
        v16h q0 = *(const v16h*)(bS0 + kb);
        v16h q1 = *(const v16h*)(bS1 + kb);

        // ---- 8 WMMAs per K-step ----
        accL[0][0] = __builtin_amdgcn_wmma_f32_16x16x32_f16(false, a0, false, b0, (short)0, accL[0][0], false, false);
        accL[0][1] = __builtin_amdgcn_wmma_f32_16x16x32_f16(false, a0, false, b1, (short)0, accL[0][1], false, false);
        accL[1][0] = __builtin_amdgcn_wmma_f32_16x16x32_f16(false, a1, false, b0, (short)0, accL[1][0], false, false);
        accL[1][1] = __builtin_amdgcn_wmma_f32_16x16x32_f16(false, a1, false, b1, (short)0, accL[1][1], false, false);
        accP[0][0] = __builtin_amdgcn_wmma_f32_16x16x32_f16(false, p0, false, q0, (short)0, accP[0][0], false, false);
        accP[0][1] = __builtin_amdgcn_wmma_f32_16x16x32_f16(false, p0, false, q1, (short)0, accP[0][1], false, false);
        accP[1][0] = __builtin_amdgcn_wmma_f32_16x16x32_f16(false, p1, false, q0, (short)0, accP[1][0], false, false);
        accP[1][1] = __builtin_amdgcn_wmma_f32_16x16x32_f16(false, p1, false, q1, (short)0, accP[1][1], false, false);
    }

    // ---- Fused flipout epilogue ----
    // C/D layout: VGPR r -> M = tileM + (lane>=16 ? 8 : 0) + r, N = tileN + (lane&15)
    #pragma unroll
    for (int ti = 0; ti < 2; ++ti) {
        #pragma unroll
        for (int tj = 0; tj < 2; ++tj) {
            const int n  = nBase + tj * 16 + mrow;
            const int mb = mBase + ti * 16 + khalf * 8;
            const float bl = b_loc[n];
            const float bs = bsamp[n];
            #pragma unroll
            for (int r = 0; r < 8; ++r) {
                const int m = mb + r;
                const size_t idx = (size_t)m * D_OUT + n;
                const float v = accL[ti][tj][r]
                              + (float)r1[idx] * accP[ti][tj][r]
                              + bl
                              + (float)r2[idx] * bs;
                out[idx] = v;
            }
        }
    }
}

// ---------------------------------------------------------------------------
// Host-side launcher. Workspace layout (needs ~48 MB + 8 KB):
//   xh  [BATCH*D_IN]  f16   (16 MB)
//   xsh [BATCH*D_IN]  f16   (16 MB)
//   wlT [D_OUT*D_IN]  f16   ( 8 MB)   transposed w_loc
//   wsT [D_OUT*D_IN]  f16   ( 8 MB)   transposed softplus(w_std)*eps_w
//   bsamp [D_OUT]     f32   ( 8 KB)
// ---------------------------------------------------------------------------
extern "C" void kernel_launch(void* const* d_in, const int* in_sizes, int n_in,
                              void* d_out, int out_size, void* d_ws, size_t ws_size,
                              hipStream_t stream) {
    const float* x     = (const float*)d_in[0];
    const float* w_loc = (const float*)d_in[1];
    const float* w_std = (const float*)d_in[2];
    const float* b_loc = (const float*)d_in[3];
    const float* b_std = (const float*)d_in[4];
    const float* eps_w = (const float*)d_in[5];
    const float* eps_b = (const float*)d_in[6];
    const int*   s     = (const int*)d_in[7];
    const int*   r1    = (const int*)d_in[8];
    const int*   r2    = (const int*)d_in[9];
    float* out = (float*)d_out;

    char* ws = (char*)d_ws;
    _Float16* xh    = (_Float16*)ws;
    _Float16* xsh   = xh  + (size_t)BATCH * D_IN;
    _Float16* wlT   = xsh + (size_t)BATCH * D_IN;
    _Float16* wsT   = wlT + (size_t)D_IN * D_OUT;
    float*    bsamp = (float*)(wsT + (size_t)D_IN * D_OUT);

    const unsigned nPrepX = (unsigned)(((size_t)BATCH * D_IN) / (256 * 4));

    flipout_prep_x<<<nPrepX, 256, 0, stream>>>(x, s, xh, xsh);
    dim3 gridW(D_OUT / 32, D_IN / 32);    // (64, 64)
    flipout_prep_w<<<gridW, 256, 0, stream>>>(w_loc, w_std, eps_w, wlT, wsT);
    flipout_prep_b<<<D_OUT / 256, 256, 0, stream>>>(b_std, eps_b, bsamp);

    dim3 grid(D_OUT / 64, BATCH / 128);   // (32, 32)
    flipout_gemm_wmma<<<grid, 256, 0, stream>>>(xh, xsh, wlT, wsT,
                                                r1, r2, b_loc, bsamp, out);
}
